// UNetCrossAttention_13537736917678
// MI455X (gfx1250) — matrix operand order repaired and checked by reading, not compiled, not run on hardware
//
#include <hip/hip_runtime.h>

#define CH    640
#define XD    768
#define NH    8
#define HD    80
#define INNER 640
#define BATCH 16
#define SEQ   4096
#define TKV   77

typedef unsigned short u16;
typedef __attribute__((ext_vector_type(16))) __bf16 bf16x16;
typedef __attribute__((ext_vector_type(8)))  float  f32x8;
typedef __attribute__((ext_vector_type(8)))  u16    u16x8;
typedef __attribute__((ext_vector_type(4)))  unsigned int u32x4;
typedef __attribute__((ext_vector_type(8)))  int    i32x8;
typedef __attribute__((ext_vector_type(4)))  int    i32x4;

union FragAB { bf16x16 v; u16x8 h[2]; };

__device__ __forceinline__ u16 f2bf(float f) {
  unsigned u = __float_as_uint(f);
  u += 0x7FFFu + ((u >> 16) & 1u);          // round-to-nearest-even
  return (u16)(u >> 16);
}

__device__ __forceinline__ f32x8 zero8() {
  f32x8 z = {0.f, 0.f, 0.f, 0.f, 0.f, 0.f, 0.f, 0.f};
  return z;
}

__device__ __forceinline__ f32x8 wmma_bf16(const FragAB& a, const FragAB& b, f32x8 c) {
  return __builtin_amdgcn_wmma_f32_16x16x32_bf16(false, a.v, false, b.v,
                                                 (short)0, c, false, false);
}

// ---------------------------------------------------------------------------
// TDM: DMA a [16 rows x 32 cols] bf16 slice of a row-major [rows x K] tensor
// into LDS at lds_off, with LDS padding of 4 DWORDs (8 u16) every 16 DWORDs
// (32 u16) -> padded row stride of 40 u16 == ASTR. Rows beyond `mrem`
// zero-fill via tensor_dim1 OOB handling. Tracked by TENSORcnt.
// D# layout per CDNA5 ISA 8.3/8.4 (count=1, type=2, data_size=2B).
// This toolchain exposes the 6-arg builtin:
//   (u32x4 g0, i32x8 g1, i32x4 g2, i32x4 g3, i32x8 extra, i32 cpol)
// ---------------------------------------------------------------------------
#define BM 128
#define BN 128
#define BK 32
#define ASTR 40   // BK+8 ushorts: 80B row stride, multiple of 16B
#define BSTR 40

__device__ __forceinline__ void tdm_load_a_slice(const u16* gptr, unsigned lds_off,
                                                 int K, int mrem) {
  unsigned long long ga = (unsigned long long)gptr;
  u32x4 g0;
  g0[0] = 1u;                                   // count=1, user mode
  g0[1] = lds_off;                              // lds_addr[31:0]
  g0[2] = (unsigned)ga;                         // global_addr[31:0]
  g0[3] = (unsigned)(ga >> 32) | (2u << 30);    // global_addr[56:32] | type=2
  i32x8 g1;
  g1[0] = (1 << 16)      // data_size = 2 bytes
        | (1 << 20)      // pad_enable
        | (3 << 22)      // pad_interval: 16 DWORDs (=32 u16 row)
        | (3 << 25);     // pad_amount: 4 DWORDs (=8 u16)
  g1[1] = (K & 0xFFFF) << 16;                            // tensor_dim0 lo16
  g1[2] = ((unsigned)K >> 16) | ((mrem & 0xFFFF) << 16); // dim0 hi | dim1 lo
  g1[3] = ((unsigned)mrem >> 16) | (BK << 16);           // dim1 hi | tile_dim0=32
  g1[4] = 16;                                            // tile_dim1 = 16 rows
  g1[5] = K;                                             // tensor_dim0_stride lo32
  g1[6] = 0;
  g1[7] = 0;
  i32x4 gz  = {0, 0, 0, 0};
  i32x8 gz8 = {0, 0, 0, 0, 0, 0, 0, 0};
  __builtin_amdgcn_tensor_load_to_lds(g0, g1, gz, gz, gz8, 0);
}

// ---------------------------------------------------------------------------
// Kernel 1: LayerNorm over channels, [B,C,S] -> bf16 [B,S,C]
// ---------------------------------------------------------------------------
__global__ __launch_bounds__(256)
void ln_to_bf16_kernel(const float* __restrict__ hidden,
                       const float* __restrict__ ln_w,
                       const float* __restrict__ ln_b,
                       u16* __restrict__ xn) {
  __shared__ float psum[4][64];
  __shared__ float psq[4][64];
  __shared__ float smu[64];
  __shared__ float srs[64];
  const int b  = blockIdx.y;
  const int s0 = blockIdx.x * 64;
  const int tx = threadIdx.x & 63;
  const int ty = threadIdx.x >> 6;
  const float* base = hidden + (size_t)b * CH * SEQ + s0 + tx;

  float sum = 0.f, sq = 0.f;
  for (int c = ty; c < CH; c += 4) {
    float v = base[(size_t)c * SEQ];
    sum += v; sq += v * v;
  }
  psum[ty][tx] = sum; psq[ty][tx] = sq;
  __syncthreads();
  if (ty == 0) {
    float s = psum[0][tx] + psum[1][tx] + psum[2][tx] + psum[3][tx];
    float q = psq[0][tx] + psq[1][tx] + psq[2][tx] + psq[3][tx];
    float mu  = s * (1.0f / CH);
    float var = q * (1.0f / CH) - mu * mu;
    smu[tx] = mu;
    srs[tx] = rsqrtf(var + 1e-6f);
  }
  __syncthreads();
  const float mu = smu[tx], rs = srs[tx];
  u16* orow = xn + (size_t)(b * SEQ + s0 + tx) * CH;
  for (int c = ty; c < CH; c += 4) {
    float v = base[(size_t)c * SEQ];
    orow[c] = f2bf((v - mu) * rs * ln_w[c] + ln_b[c]);
  }
}

// ---------------------------------------------------------------------------
// Kernel 2: f32 -> bf16 elementwise
// ---------------------------------------------------------------------------
__global__ __launch_bounds__(256)
void f32_to_bf16_kernel(const float* __restrict__ src, u16* __restrict__ dst, int n) {
  int i = blockIdx.x * 256 + threadIdx.x;
  if (i < n) dst[i] = f2bf(src[i]);
}

// ---------------------------------------------------------------------------
// Kernel 3: bf16 WMMA GEMM core, TDM-fed A tiles, double-buffered LDS.
// Block 256 thr (8 waves), tile 128x128, K-step 32, one barrier per K-step.
// EPI=0: C[M,N] bf16 store (M guarded).
// EPI=1: out = acc + bo[n] + residual (transposed [B,C,S] scatter), f32.
// ---------------------------------------------------------------------------
template <int EPI>
__global__ __launch_bounds__(256)
void gemm_core(const u16* __restrict__ A, const u16* __restrict__ Bw,
               u16* __restrict__ Cbf, const float* __restrict__ bo,
               const float* __restrict__ residual, float* __restrict__ outf,
               int M, int N, int K) {
  __shared__ u16 As[2][BM][ASTR];
  __shared__ u16 Bs[2][BN][BSTR];   // transposed: Bs[p][n][k]
  const int tid  = threadIdx.x;
  const int lane = tid & 31;
  const int wave = tid >> 5;
  const int wr   = wave >> 1;          // 0..3  (M sub-block of 32)
  const int wc   = wave & 1;           // 0..1  (N sub-block of 64)
  const int m_blk = blockIdx.y * BM;
  const int n_blk = blockIdx.x * BN;
  const int half = lane >> 4;
  const int l16  = lane & 15;

  f32x8 acc[2][4];
#pragma unroll
  for (int i = 0; i < 2; ++i)
#pragma unroll
    for (int j = 0; j < 4; ++j) acc[i][j] = zero8();

  const int b_k = tid & 31;
  const int b_n = (tid >> 5) * 16;

  // Per-wave TDM slice: rows [m_blk + wave*16, +16) of A.
  const int arow0 = m_blk + wave * 16;
  int mrem = M - arow0; if (mrem < 0) mrem = 0;
  const unsigned lds_a0 = (unsigned)(size_t)(&As[0][0][0]);
  const unsigned slice_off = (unsigned)(wave * 16 * ASTR * 2);
  const unsigned buf_stride = (unsigned)(BM * ASTR * 2);

  // ---- prologue: stage 0 ----
  tdm_load_a_slice(A + (size_t)arow0 * K, lds_a0 + slice_off, K, mrem);
  {
    const u16* src = Bw + (size_t)b_k * N + n_blk + b_n;
    uint4 r0 = *(const uint4*)(src);
    uint4 r1 = *(const uint4*)(src + 8);
    u16 tmp[16];
    *(uint4*)(tmp) = r0; *(uint4*)(tmp + 8) = r1;
#pragma unroll
    for (int e = 0; e < 16; ++e) Bs[0][b_n + e][b_k] = tmp[e];
  }
  __builtin_amdgcn_s_wait_tensorcnt(0);
  __syncthreads();

  int p = 0;
  for (int k0 = 0; k0 < K; k0 += BK) {
    const int np = p ^ 1;
    const bool nxt = (k0 + BK) < K;
    uint4 br0, br1;
    if (nxt) {
      // async DMA of next A tile into the idle buffer (safe: nobody reads np)
      tdm_load_a_slice(A + (size_t)arow0 * K + (k0 + BK),
                       lds_a0 + (unsigned)np * buf_stride + slice_off, K, mrem);
      // next B tile global->regs (overlaps WMMA below)
      const u16* src = Bw + (size_t)(k0 + BK + b_k) * N + n_blk + b_n;
      br0 = *(const uint4*)(src);
      br1 = *(const uint4*)(src + 8);
    }

    // ---- compute current buffer ----
    FragAB af[2], bf[4];
#pragma unroll
    for (int mi = 0; mi < 2; ++mi) {
      const u16* ptr = &As[p][wr * 32 + mi * 16 + l16][half * 8];
      af[mi].h[0] = *(const u16x8*)(ptr);
      af[mi].h[1] = *(const u16x8*)(ptr + 16);
    }
#pragma unroll
    for (int ni = 0; ni < 4; ++ni) {
      const u16* ptr = &Bs[p][wc * 64 + ni * 16 + l16][half * 8];
      bf[ni].h[0] = *(const u16x8*)(ptr);
      bf[ni].h[1] = *(const u16x8*)(ptr + 16);
    }
#pragma unroll
    for (int mi = 0; mi < 2; ++mi)
#pragma unroll
      for (int ni = 0; ni < 4; ++ni)
        acc[mi][ni] = wmma_bf16(af[mi], bf[ni], acc[mi][ni]);

    if (nxt) {
      u16 tmp[16];
      *(uint4*)(tmp) = br0; *(uint4*)(tmp + 8) = br1;
#pragma unroll
      for (int e = 0; e < 16; ++e) Bs[np][b_n + e][b_k] = tmp[e];
      __builtin_amdgcn_s_wait_tensorcnt(0);
    }
    __syncthreads();
    p = np;
  }

  // ---- epilogue ----
#pragma unroll
  for (int mi = 0; mi < 2; ++mi)
#pragma unroll
    for (int ni = 0; ni < 4; ++ni)
#pragma unroll
      for (int r = 0; r < 8; ++r) {
        int gm = m_blk + wr * 32 + mi * 16 + r + half * 8;
        int gn = n_blk + wc * 64 + ni * 16 + l16;
        if (EPI == 0) {
          if (gm < M) Cbf[(size_t)gm * N + gn] = f2bf(acc[mi][ni][r]);
        } else {
          int bb = gm >> 12;          // / SEQ
          int ss = gm & (SEQ - 1);
          size_t idx = (size_t)(bb * CH + gn) * SEQ + ss;
          outf[idx] = acc[mi][ni][r] + bo[gn] + residual[idx];
        }
      }
}

// ---------------------------------------------------------------------------
// Kernel 4: fused cross-attention for one (b, h, 64-row s-tile).
// 128 threads = 4 waves; each wave owns 16 q rows.
// HD=80 padded to K=96; T=77 padded to 80 (N) / 96 (K).
// ---------------------------------------------------------------------------
__global__ __launch_bounds__(128)
void attention_kernel(const u16* __restrict__ q, const u16* __restrict__ k,
                      const u16* __restrict__ v, u16* __restrict__ o) {
  __shared__ union {
    struct { u16 qs[64][104]; u16 ks[80][104]; } a;   // 13312 + 16640 B
    float scores[64][80];                             // 20480 B (reuses dead qs/ks)
  } sm;
  __shared__ u16 vts[80][104];    // vts[d][t]
  __shared__ u16 attns[64][104];  // softmaxed probs, K-padded to 96

  const int b  = blockIdx.z;
  const int h  = blockIdx.y;
  const int s0 = blockIdx.x * 64;
  const int tid = threadIdx.x;

  for (int idx = tid; idx < 64 * 96; idx += 128) {
    int row = idx / 96, d = idx % 96;
    sm.a.qs[row][d] = (d < HD)
        ? q[(size_t)(b * SEQ + s0 + row) * INNER + h * HD + d] : (u16)0;
  }
  for (int idx = tid; idx < 80 * 96; idx += 128) {
    int t = idx / 96, d = idx % 96;
    sm.a.ks[t][d] = (t < TKV && d < HD)
        ? k[(size_t)(b * TKV + t) * INNER + h * HD + d] : (u16)0;
  }
  for (int idx = tid; idx < 80 * 96; idx += 128) {
    int d = idx / 96, t = idx % 96;
    vts[d][t] = (t < TKV)
        ? v[(size_t)(b * TKV + t) * INNER + h * HD + d] : (u16)0;
  }
  __syncthreads();

  const int lane = tid & 31;
  const int wave = tid >> 5;
  const int half = lane >> 4;
  const int l16  = lane & 15;
  const int m0   = wave * 16;

  // --- scores = q @ k^T ---
  f32x8 acc[5];
#pragma unroll
  for (int ni = 0; ni < 5; ++ni) acc[ni] = zero8();
#pragma unroll
  for (int kc = 0; kc < 3; ++kc) {
    FragAB af;
    const u16* ap = &sm.a.qs[m0 + l16][kc * 32 + half * 8];
    af.h[0] = *(const u16x8*)(ap);
    af.h[1] = *(const u16x8*)(ap + 16);
#pragma unroll
    for (int ni = 0; ni < 5; ++ni) {
      FragAB bfr;
      const u16* bp = &sm.a.ks[ni * 16 + l16][kc * 32 + half * 8];
      bfr.h[0] = *(const u16x8*)(bp);
      bfr.h[1] = *(const u16x8*)(bp + 16);
      acc[ni] = wmma_bf16(af, bfr, acc[ni]);
    }
  }
  __syncthreads();   // qs/ks reads done before scores aliases them

  const float scale = 0.11180339887498949f;  // 1/sqrt(80)
#pragma unroll
  for (int ni = 0; ni < 5; ++ni)
#pragma unroll
    for (int r = 0; r < 8; ++r)
      sm.scores[m0 + r + half * 8][ni * 16 + l16] = acc[ni][r] * scale;
  __syncthreads();

  // --- softmax over T (one thread per row) ---
  if (tid < 64) {
    float mx = -1e30f;
    for (int t = 0; t < TKV; ++t) mx = fmaxf(mx, sm.scores[tid][t]);
    float sumv = 0.f;
    for (int t = 0; t < TKV; ++t) sumv += __expf(sm.scores[tid][t] - mx);
    float inv = 1.0f / sumv;
    for (int t = 0; t < 96; ++t)
      attns[tid][t] = (t < TKV) ? f2bf(__expf(sm.scores[tid][t] - mx) * inv)
                                : (u16)0;
  }
  __syncthreads();

  // --- out = attn @ v ---
  f32x8 acc2[5];
#pragma unroll
  for (int ni = 0; ni < 5; ++ni) acc2[ni] = zero8();
#pragma unroll
  for (int kc = 0; kc < 3; ++kc) {
    FragAB af;
    const u16* ap = &attns[m0 + l16][kc * 32 + half * 8];
    af.h[0] = *(const u16x8*)(ap);
    af.h[1] = *(const u16x8*)(ap + 16);
#pragma unroll
    for (int ni = 0; ni < 5; ++ni) {
      FragAB bfr;
      const u16* bp = &vts[ni * 16 + l16][kc * 32 + half * 8];
      bfr.h[0] = *(const u16x8*)(bp);
      bfr.h[1] = *(const u16x8*)(bp + 16);
      acc2[ni] = wmma_bf16(af, bfr, acc2[ni]);
    }
  }

#pragma unroll
  for (int ni = 0; ni < 5; ++ni)
#pragma unroll
    for (int r = 0; r < 8; ++r) {
      int m = m0 + r + half * 8;
      int d = ni * 16 + l16;   // < 80
      o[(size_t)(b * SEQ + s0 + m) * INNER + h * HD + d] = f2bf(acc2[ni][r]);
    }
}

// ---------------------------------------------------------------------------
extern "C" void kernel_launch(void* const* d_in, const int* in_sizes, int n_in,
                              void* d_out, int out_size, void* d_ws, size_t ws_size,
                              hipStream_t stream) {
  const float* hidden = (const float*)d_in[0];
  const float* enc    = (const float*)d_in[1];
  const float* ln_w   = (const float*)d_in[2];
  const float* ln_b   = (const float*)d_in[3];
  const float* wq     = (const float*)d_in[4];
  const float* wk     = (const float*)d_in[5];
  const float* wv     = (const float*)d_in[6];
  const float* wo     = (const float*)d_in[7];
  const float* bo     = (const float*)d_in[8];
  float* out = (float*)d_out;

  char* ws = (char*)d_ws;
  size_t off = 0;
  auto alloc = [&](size_t bytes) -> char* {
    char* p = ws + off;
    off = (off + bytes + 255) & ~(size_t)255;
    return p;
  };

  const int M   = BATCH * SEQ;       // 65536
  const int Mk  = BATCH * TKV;       // 1232
  const int MkP = 1280;              // Mk padded to BM multiple (TDM over-read safe)
  u16* xn   = (u16*)alloc((size_t)M * INNER * 2);    // reused as attn output
  u16* qb   = (u16*)alloc((size_t)M * INNER * 2);
  u16* encb = (u16*)alloc((size_t)MkP * XD * 2);
  u16* kb   = (u16*)alloc((size_t)Mk * INNER * 2);
  u16* vb   = (u16*)alloc((size_t)Mk * INNER * 2);
  u16* wqb  = (u16*)alloc((size_t)CH * INNER * 2);
  u16* wkb  = (u16*)alloc((size_t)XD * INNER * 2);
  u16* wvb  = (u16*)alloc((size_t)XD * INNER * 2);
  u16* wob  = (u16*)alloc((size_t)INNER * CH * 2);
  (void)ws_size; (void)in_sizes; (void)n_in; (void)out_size;

  auto cvt = [&](const float* s, u16* d, int n) {
    f32_to_bf16_kernel<<<(n + 255) / 256, 256, 0, stream>>>(s, d, n);
  };
  cvt(wq, wqb, CH * INNER);
  cvt(wk, wkb, XD * INNER);
  cvt(wv, wvb, XD * INNER);
  cvt(wo, wob, INNER * CH);
  cvt(enc, encb, Mk * XD);

  ln_to_bf16_kernel<<<dim3(SEQ / 64, BATCH), 256, 0, stream>>>(hidden, ln_w, ln_b, xn);

  // q = xn @ wq      [65536 x 640] = [65536 x 640][640 x 640]
  gemm_core<0><<<dim3(INNER / BN, M / BM), 256, 0, stream>>>(
      xn, wqb, qb, nullptr, nullptr, nullptr, M, INNER, CH);
  // k = enc @ wk, v = enc @ wv   [1232 x 640] = [1232 x 768][768 x 640]
  gemm_core<0><<<dim3(INNER / BN, MkP / BM), 256, 0, stream>>>(
      encb, wkb, kb, nullptr, nullptr, nullptr, Mk, INNER, XD);
  gemm_core<0><<<dim3(INNER / BN, MkP / BM), 256, 0, stream>>>(
      encb, wvb, vb, nullptr, nullptr, nullptr, Mk, INNER, XD);

  // fused attention; writes bf16 attn-out into xn (xn dead after q projection)
  attention_kernel<<<dim3(SEQ / 64, NH, BATCH), 128, 0, stream>>>(qb, kb, vb, xn);

  // out = attn_out @ wo + bo + residual, transposed back to [B,C,H,W]
  gemm_core<1><<<dim3(CH / BN, M / BM), 256, 0, stream>>>(
      xn, wob, nullptr, bo, hidden, out, M, CH, INNER);
}
